// CeilingSymbolicRNN_3917010174116
// MI455X (gfx1250) — compile-verified, matrix operand order checked by reading, not run on hardware
//
#include <hip/hip_runtime.h>
#include <hip/hip_bf16.h>

// ---------------------------------------------------------------------------
// Problem constants (from reference)
// ---------------------------------------------------------------------------
#define Mrows   8192          // B*S = 4*2048
#define Dd      512
#define TWO_D   1024
#define Vv      16000
#define NSYM    32
#define NCON    8
#define MAXD    6
#define LOOK    3
#define EPS_F   1e-8f
#define COMMIT_F 0.25f
#define SCALE_F 0.04419417382415922f   // 512^-0.5

// ---------------------------------------------------------------------------
// Types
// ---------------------------------------------------------------------------
typedef __attribute__((ext_vector_type(16))) __bf16          v16bf;
typedef __attribute__((ext_vector_type(16))) unsigned short  v16u16;
typedef __attribute__((ext_vector_type(8)))  unsigned short  v8u16;
typedef __attribute__((ext_vector_type(8)))  float           v8f;
typedef __attribute__((ext_vector_type(4)))  int             v4i;

__device__ __forceinline__ unsigned short f32_to_bf16(float f) {
    unsigned int u = __float_as_uint(f);
    u += 0x7FFFu + ((u >> 16) & 1u);     // round-to-nearest-even
    return (unsigned short)(u >> 16);
}

__device__ __forceinline__ float wave_reduce_add(float v) {
#pragma unroll
    for (int off = 16; off > 0; off >>= 1) v += __shfl_xor(v, off, 32);
    return v;
}

// ---------------------------------------------------------------------------
// gfx1250 async global->LDS copy (ASYNCcnt path), with safe fallback.
// Verified signature: (AS1 v4i*, AS3 v4i*, i32 offset, i32 cpol)
// ---------------------------------------------------------------------------
#if defined(__has_builtin)
# if __has_builtin(__builtin_amdgcn_global_load_async_to_lds_b128)
#  define HAVE_ASYNC_LDS 1
# endif
#endif
#ifndef HAVE_ASYNC_LDS
# define HAVE_ASYNC_LDS 0
#endif

typedef __attribute__((address_space(1))) v4i* gvec_ptr;   // pointer to AS1 v4i
typedef __attribute__((address_space(3))) v4i* lvec_ptr;   // pointer to AS3 v4i

__device__ __forceinline__ void async_copy16(const unsigned short* g, unsigned short* l) {
#if HAVE_ASYNC_LDS
    __builtin_amdgcn_global_load_async_to_lds_b128(
        (gvec_ptr)(unsigned long long)g,
        (lvec_ptr)(unsigned int)(unsigned long long)l,
        0, 0);
#else
    *(v8u16*)l = *(const v8u16*)g;
#endif
}

__device__ __forceinline__ void async_wait_all() {
#if HAVE_ASYNC_LDS
# if __has_builtin(__builtin_amdgcn_s_wait_asynccnt)
    __builtin_amdgcn_s_wait_asynccnt(0);
# else
    asm volatile("s_wait_asynccnt 0x0" ::: "memory");
# endif
#endif
}

// ---------------------------------------------------------------------------
// bf16 WMMA GEMM:  C[M,N] = A[M,K] @ W[N,K]^T + bias
//   A: bf16 bits row-major (lda in elements); W: bf16 bits row-major [N,K]
// Tiles 128x64x32, 128 threads = 4 waves; each wave computes 32x64
// (two A fragments sharing four B fragments -> 8 WMMAs per K-step).
// Double-buffered LDS; one barrier per K-step; staging = async b128 DMA.
// ---------------------------------------------------------------------------
#define BM 128
#define BN 64
#define BK 32
#define BKP 40   // padded LDS row stride (ushorts); 80B keeps 16B alignment

__launch_bounds__(128)
__global__ void gemm_bf16_wmma(const unsigned short* __restrict__ A, int lda,
                               const unsigned short* __restrict__ Wb,
                               int K, int N,
                               float* __restrict__ C, int ldc,
                               const float* __restrict__ bias)
{
    __shared__ unsigned short sA[2][BM][BKP];
    __shared__ unsigned short sB[2][BN][BKP];

    const int tid  = threadIdx.x;
    const int wave = tid >> 5;
    const int lane = tid & 31;
    const int bm   = blockIdx.y * BM;
    const int bn   = blockIdx.x * BN;
    const int l    = lane & 15;
    const int hi   = lane >> 4;

    const int KT = K / BK;

    // staging: A tile 128x32 = 512 octets (4/thread), B tile 64x32 = 256 (2/thread)
    auto stage = [&](int buf, int k0) {
#pragma unroll
        for (int i = 0; i < 4; ++i) {
            int o = tid + i * 128;
            int r = o >> 2, c = (o & 3) * 8;
            async_copy16(A + (size_t)(bm + r) * lda + k0 + c, &sA[buf][r][c]);
        }
#pragma unroll
        for (int i = 0; i < 2; ++i) {
            int o = tid + i * 128;
            int r = o >> 2, c = (o & 3) * 8;
            async_copy16(Wb + (size_t)(bn + r) * K + k0 + c, &sB[buf][r][c]);
        }
    };

    v8f acc[2][4] = {};

    stage(0, 0);
    for (int kt = 0; kt < KT; ++kt) {
        const int cur = kt & 1;
        async_wait_all();        // my stage(kt) is complete
        __syncthreads();         // everyone's stage(kt) visible; prev compute done
        if (kt + 1 < KT) stage(cur ^ 1, (kt + 1) * BK);   // overlaps with WMMAs

        // two A fragments (rows wave*32 .. +16 .. +32); ISA 16x32 bf16 layout
        v16bf a[2];
#pragma unroll
        for (int s = 0; s < 2; ++s) {
            const unsigned short* ap = &sA[cur][wave * 32 + s * 16 + l][0];
            v8u16 a0 = *(const v8u16*)(ap + hi * 8);
            v8u16 a1 = *(const v8u16*)(ap + hi * 8 + 16);
            a[s] = __builtin_bit_cast(v16bf, __builtin_shufflevector(a0, a1,
                     0,1,2,3,4,5,6,7,8,9,10,11,12,13,14,15));
        }
        v16bf b[4];
#pragma unroll
        for (int j = 0; j < 4; ++j) {
            const unsigned short* bp = &sB[cur][j * 16 + l][0];
            v8u16 b0 = *(const v8u16*)(bp + hi * 8);
            v8u16 b1 = *(const v8u16*)(bp + hi * 8 + 16);
            b[j] = __builtin_bit_cast(v16bf, __builtin_shufflevector(b0, b1,
                     0,1,2,3,4,5,6,7,8,9,10,11,12,13,14,15));
        }
#pragma unroll
        for (int s = 0; s < 2; ++s)
#pragma unroll
            for (int j = 0; j < 4; ++j)
                acc[s][j] = __builtin_amdgcn_wmma_f32_16x16x32_bf16(
                              false, a[s], false, b[j], (short)0, acc[s][j], false, false);
    }

    // write back: VGPR r -> (M = wave*32 + s*16 + hi*8 + r, N = j*16 + l)
#pragma unroll
    for (int j = 0; j < 4; ++j) {
        int n = bn + j * 16 + l;
        float bvv = bias ? bias[n] : 0.0f;
#pragma unroll
        for (int s = 0; s < 2; ++s)
#pragma unroll
            for (int r = 0; r < 8; ++r) {
                int m = bm + wave * 32 + s * 16 + hi * 8 + r;
                C[(size_t)m * ldc + n] = acc[s][j][r] + bvv;
            }
    }
}

// ---------------------------------------------------------------------------
// Weight prep
// ---------------------------------------------------------------------------
__global__ void convert_bf16_kernel(const float* __restrict__ in,
                                    unsigned short* __restrict__ out, int n)
{
    int i = blockIdx.x * blockDim.x + threadIdx.x;
    if (i < n) out[i] = f32_to_bf16(in[i]);
}

// Pack complex-cell weights: Wcr = [Wr | -Wi], Wci = [Wi | Wr]  (rows of N=512)
__global__ void pack_cell_weights(const float* __restrict__ Wr,
                                  const float* __restrict__ Wi,
                                  unsigned short* __restrict__ Wcr,
                                  unsigned short* __restrict__ Wci)
{
    int i = blockIdx.x * blockDim.x + threadIdx.x;   // over 512*1024
    if (i >= Dd * TWO_D) return;
    int n = i / TWO_D, k = i % TWO_D;
    float vr, vi;
    if (k < Dd) { vr = Wr[(size_t)n * Dd + k];          vi = Wi[(size_t)n * Dd + k]; }
    else        { vr = -Wi[(size_t)n * Dd + (k - Dd)];  vi = Wr[(size_t)n * Dd + (k - Dd)]; }
    Wcr[i] = f32_to_bf16(vr);
    Wci[i] = f32_to_bf16(vi);
}

// ---------------------------------------------------------------------------
// Embedding: zf/zb[m, 0:D] = r*cos(th), [D:2D] = r*sin(th)
// ---------------------------------------------------------------------------
__global__ void embed_kernel(const int* __restrict__ ids,
                             const float* __restrict__ emb_mag,
                             const float* __restrict__ emb_phase,
                             float* __restrict__ zf, unsigned short* __restrict__ zb)
{
    int i = blockIdx.x * blockDim.x + threadIdx.x;      // over Mrows*Dd
    if (i >= Mrows * Dd) return;
    int m = i / Dd, d = i % Dd;
    int id = ids[m];
    float r  = emb_mag[(size_t)id * Dd + d];
    float th = emb_phase[(size_t)id * Dd + d];
    float zr = r * cosf(th), zi = r * sinf(th);
    size_t i0 = (size_t)m * TWO_D + d, i1 = i0 + Dd;
    zf[i0] = zr; zb[i0] = f32_to_bf16(zr);
    zf[i1] = zi; zb[i1] = f32_to_bf16(zi);
}

// ---------------------------------------------------------------------------
// Cell nonlinearity: z = tanh(o / (1 + sqrt(or^2 + oi^2 + eps)))
// ---------------------------------------------------------------------------
__global__ void cell_post_kernel(const float* __restrict__ of,
                                 float* __restrict__ zf, unsigned short* __restrict__ zb)
{
    int i = blockIdx.x * blockDim.x + threadIdx.x;      // over Mrows*Dd
    if (i >= Mrows * Dd) return;
    int m = i / Dd, d = i % Dd;
    size_t i0 = (size_t)m * TWO_D + d, i1 = i0 + Dd;
    float o_r = of[i0], o_i = of[i1];
    float inv = 1.0f / (1.0f + sqrtf(o_r * o_r + o_i * o_i + EPS_F));
    float a = tanhf(o_r * inv), b = tanhf(o_i * inv);
    zf[i0] = a; zb[i0] = f32_to_bf16(a);
    zf[i1] = b; zb[i1] = f32_to_bf16(b);
}

// ---------------------------------------------------------------------------
// Fused attention (one wave per row, T <= 5)
// ---------------------------------------------------------------------------
__global__ void attn_kernel(float* __restrict__ zf, unsigned short* __restrict__ zb,
                            const float* __restrict__ Q,
                            const float* __restrict__ Khist,
                            const float* __restrict__ Vhist,
                            const float* __restrict__ conf, int T)
{
    int wave = threadIdx.x >> 5, lane = threadIdx.x & 31;
    int m = blockIdx.x * 8 + wave;
    if (m >= Mrows) return;

    const float* q = Q + (size_t)m * Dd;
    float cf = conf[m];
    float s[5];
    for (int t = 0; t < T; ++t) {
        const float* k = Khist + ((size_t)t * Mrows + m) * Dd;
        float p = 0.0f;
        for (int d = lane; d < Dd; d += 32) p += q[d] * k[d];
        s[t] = wave_reduce_add(p) * SCALE_F * cf;
    }
    float mx = s[0];
    for (int t = 1; t < T; ++t) mx = fmaxf(mx, s[t]);
    float sum = 0.0f;
    for (int t = 0; t < T; ++t) { s[t] = expf(s[t] - mx); sum += s[t]; }
    float inv = 1.0f / sum;
    for (int d = lane; d < TWO_D; d += 32) {
        float c = 0.0f;
        for (int t = 0; t < T; ++t)
            c += s[t] * Vhist[((size_t)t * Mrows + m) * TWO_D + d];
        size_t idx = (size_t)m * TWO_D + d;
        float nv = zf[idx] + 0.1f * (c * inv);
        zf[idx] = nv; zb[idx] = f32_to_bf16(nv);
    }
}

// ---------------------------------------------------------------------------
// Setup: zero loss accumulators; codebook norms; min con-distance table
// ---------------------------------------------------------------------------
__global__ void setup_kernel(const float* __restrict__ sym_cb,
                             const float* __restrict__ con_cb,
                             float* __restrict__ sym_norm,
                             float* __restrict__ con_min,
                             float* __restrict__ acc)
{
    int s = threadIdx.x;                 // 32 threads
    if (s == 0) { acc[0] = 0.0f; acc[1] = 0.0f; acc[2] = 0.0f; }
    const float* sp = sym_cb + (size_t)s * TWO_D;
    float sn = 0.0f;
    for (int d = 0; d < TWO_D; ++d) sn += sp[d] * sp[d];
    sym_norm[s] = sn;
    float best = 3.4e38f;
    for (int c = 0; c < NCON; ++c) {
        const float* cp = con_cb + (size_t)c * TWO_D;
        float cn = 0.0f, dt = 0.0f;
        for (int d = 0; d < TWO_D; ++d) { cn += cp[d] * cp[d]; dt += sp[d] * cp[d]; }
        best = fminf(best, sn + cn - 2.0f * dt);
    }
    con_min[s] = best;
}

// ---------------------------------------------------------------------------
// VQ: one wave per row; lane c owns code c (NSYM == 32 == wave width)
// ---------------------------------------------------------------------------
__global__ void quantize_kernel(float* __restrict__ zf, unsigned short* __restrict__ zb,
                                const float* __restrict__ sym_cb,
                                const float* __restrict__ sym_norm,
                                const float* __restrict__ con_min,
                                float* __restrict__ conf,
                                int* __restrict__ prev_sidx,
                                float* __restrict__ acc, int do_gain)
{
    int wave = threadIdx.x >> 5, lane = threadIdx.x & 31;
    int m = blockIdx.x * 8 + wave;
    if (m >= Mrows) return;

    float* z = zf + (size_t)m * TWO_D;

    float zz = 0.0f;
    for (int d = lane; d < TWO_D; d += 32) zz += z[d] * z[d];
    zz = wave_reduce_add(zz);

    const float* cb = sym_cb + (size_t)lane * TWO_D;   // lane == code index
    float dot = 0.0f;
    for (int d = 0; d < TWO_D; ++d) dot += z[d] * cb[d];   // z broadcast across lanes
    float dist = zz + sym_norm[lane] - 2.0f * dot;

    int best = lane;
#pragma unroll
    for (int off = 16; off > 0; off >>= 1) {
        float od = __shfl_xor(dist, off, 32);
        int   ob = __shfl_xor(best, off, 32);
        if (od < dist || (od == dist && ob < best)) { dist = od; best = ob; }
    }
    int   sidx = best;
    float mind = dist;

    if (lane == 0) {
        conf[m] = 1.0f / (1.0f + mind);
        const float inv_el = 1.0f / ((float)Mrows * (float)TWO_D);
        atomicAdd(&acc[0], (1.0f + COMMIT_F) * mind * inv_el);
        atomicAdd(&acc[1], (1.0f + COMMIT_F) * con_min[sidx] * inv_el);
        if (do_gain && sidx != prev_sidx[m])
            atomicAdd(&acc[2], logf((1.0f + EPS_F) / EPS_F) / (float)Mrows);
        prev_sidx[m] = sidx;
    }

    const float* w = sym_cb + (size_t)sidx * TWO_D;
    for (int d = lane; d < TWO_D; d += 32) {
        float nv = 0.5f * z[d] + 0.5f * w[d];
        z[d] = nv;
        zb[(size_t)m * TWO_D + d] = f32_to_bf16(nv);
    }
}

// ---------------------------------------------------------------------------
// Host-side launch
// ---------------------------------------------------------------------------
static inline void launch_gemm(const unsigned short* A, int lda, const unsigned short* Wb,
                               int K, int N, float* C, int ldc, const float* bias,
                               hipStream_t stream)
{
    dim3 grid(N / BN, Mrows / BM);
    gemm_bf16_wmma<<<grid, 128, 0, stream>>>(A, lda, Wb, K, N, C, ldc, bias);
}

extern "C" void kernel_launch(void* const* d_in, const int* in_sizes, int n_in,
                              void* d_out, int out_size, void* d_ws, size_t ws_size,
                              hipStream_t stream)
{
    const int*   ids       = (const int*)  d_in[0];
    const float* emb_mag   = (const float*)d_in[1];
    const float* emb_phase = (const float*)d_in[2];
    const float* Wr        = (const float*)d_in[3];
    const float* Wi        = (const float*)d_in[4];
    const float* Wq        = (const float*)d_in[5];
    const float* bq        = (const float*)d_in[6];
    const float* Wk        = (const float*)d_in[7];
    const float* bk        = (const float*)d_in[8];
    const float* Wv        = (const float*)d_in[9];
    const float* bv        = (const float*)d_in[10];
    const float* Wdec      = (const float*)d_in[11];
    const float* bdec      = (const float*)d_in[12];
    const float* sym_cb    = (const float*)d_in[13];
    const float* con_cb    = (const float*)d_in[14];

    float* logits = (float*)d_out;                       // [8192, 16000]
    float* acc    = logits + (size_t)Mrows * Vv;         // tsym, tcon, gain

    // ---- workspace carve-out (256B aligned) ----
    char* p = (char*)d_ws;
    auto carve = [&](size_t bytes) -> char* {
        char* r = p; p += (bytes + 255) & ~(size_t)255; return r;
    };
    float*          zf     = (float*)carve((size_t)Mrows * TWO_D * 4);
    unsigned short* zb     = (unsigned short*)carve((size_t)Mrows * TWO_D * 2);
    float*          of     = (float*)carve((size_t)Mrows * TWO_D * 4);
    float*          Qb     = (float*)carve((size_t)Mrows * Dd * 4);
    float*          Khist  = (float*)carve((size_t)5 * Mrows * Dd * 4);
    float*          Vhist  = (float*)carve((size_t)5 * Mrows * TWO_D * 4);
    float*          confb  = (float*)carve((size_t)Mrows * 4);
    int*            prev   = (int*)  carve((size_t)Mrows * 4);
    float*          symn   = (float*)carve(NSYM * 4);
    float*          conmin = (float*)carve(NSYM * 4);
    unsigned short* Wcr_b  = (unsigned short*)carve((size_t)Dd * TWO_D * 2);
    unsigned short* Wci_b  = (unsigned short*)carve((size_t)Dd * TWO_D * 2);
    unsigned short* Wq_b   = (unsigned short*)carve((size_t)Dd * TWO_D * 2);
    unsigned short* Wk_b   = (unsigned short*)carve((size_t)Dd * TWO_D * 2);
    unsigned short* Wv_b   = (unsigned short*)carve((size_t)TWO_D * TWO_D * 2);
    unsigned short* Wdec_b = (unsigned short*)carve((size_t)Vv * TWO_D * 2);

    // ---- setup + weight prep (once per call) ----
    setup_kernel<<<1, 32, 0, stream>>>(sym_cb, con_cb, symn, conmin, acc);
    pack_cell_weights<<<(Dd * TWO_D + 255) / 256, 256, 0, stream>>>(Wr, Wi, Wcr_b, Wci_b);
    auto conv = [&](const float* src, unsigned short* dst, int n) {
        convert_bf16_kernel<<<(n + 255) / 256, 256, 0, stream>>>(src, dst, n);
    };
    conv(Wq,   Wq_b,   Dd * TWO_D);
    conv(Wk,   Wk_b,   Dd * TWO_D);
    conv(Wv,   Wv_b,   TWO_D * TWO_D);
    conv(Wdec, Wdec_b, Vv * TWO_D);

    // ---- embedding ----
    embed_kernel<<<(Mrows * Dd + 255) / 256, 256, 0, stream>>>(ids, emb_mag, emb_phase, zf, zb);

    // ---- main loop ----
    for (int it = 0; it < MAXD; ++it) {
        // complex cell as two packed GEMMs over [zr | zi]
        launch_gemm(zb, TWO_D, Wcr_b, TWO_D, Dd, of,      TWO_D, nullptr, stream);
        launch_gemm(zb, TWO_D, Wci_b, TWO_D, Dd, of + Dd, TWO_D, nullptr, stream);
        cell_post_kernel<<<(Mrows * Dd + 255) / 256, 256, 0, stream>>>(of, zf, zb);

        if (it > 0) {   // attention over cached history
            launch_gemm(zb, TWO_D, Wq_b, TWO_D, Dd, Qb, Dd, bq, stream);
            attn_kernel<<<Mrows / 8, 256, 0, stream>>>(zf, zb, Qb, Khist, Vhist, confb, it);
        }

        quantize_kernel<<<Mrows / 8, 256, 0, stream>>>(zf, zb, sym_cb, symn, conmin,
                                                       confb, prev, acc, it > 0 ? 1 : 0);

        if (it < MAXD - 1) {   // project K/V once per appended history entry
            launch_gemm(zb, TWO_D, Wk_b, TWO_D, Dd,    Khist + (size_t)it * Mrows * Dd,    Dd,    bk, stream);
            launch_gemm(zb, TWO_D, Wv_b, TWO_D, TWO_D, Vhist + (size_t)it * Mrows * TWO_D, TWO_D, bv, stream);
        }
    }

    // ---- lookahead cells ----
    for (int it = 0; it < LOOK; ++it) {
        launch_gemm(zb, TWO_D, Wcr_b, TWO_D, Dd, of,      TWO_D, nullptr, stream);
        launch_gemm(zb, TWO_D, Wci_b, TWO_D, Dd, of + Dd, TWO_D, nullptr, stream);
        cell_post_kernel<<<(Mrows * Dd + 255) / 256, 256, 0, stream>>>(of, zf, zb);
    }

    // ---- decoder: logits = zf @ Wdec^T + bdec ----
    launch_gemm(zb, TWO_D, Wdec_b, TWO_D, Vv, logits, Vv, bdec, stream);
}
